// PCEncoder_45835890983248
// MI455X (gfx1250) — matrix operand order repaired and checked by reading, not compiled, not run on hardware
//
#include <hip/hip_runtime.h>

typedef _Float16 half_t;
typedef __attribute__((ext_vector_type(16))) _Float16 v16h;
typedef __attribute__((ext_vector_type(8)))  _Float16 v8h;
typedef __attribute__((ext_vector_type(8)))  float    v8f;

#define NB 16  // batch size (fixed by reference)

// ---------------------------------------------------------------------------
// Farthest point sampling: one block per batch, min-dist in registers,
// shared-mem argmax with first-index tie-break (matches jnp.argmax).
// Also gathers new_xyz.
// ---------------------------------------------------------------------------
__global__ __launch_bounds__(256)
void fps_kernel(const float* __restrict__ xyz, int N, int npoint,
                int* __restrict__ fidx, float* __restrict__ new_xyz)
{
  const int b = blockIdx.x;
  const int tid = threadIdx.x;
  const float* xb = xyz + (size_t)b * N * 3;
  __shared__ float s_v[256];
  __shared__ int   s_i[256];
  __shared__ float cpt[3];
  float mind[16];
  const int P = (N + 255) >> 8;
  for (int p = 0; p < P; ++p) mind[p] = 1e10f;
  int far = 0;
  for (int it = 0; it < npoint; ++it) {
    if (tid == 0) {
      fidx[b * npoint + it] = far;
      float x = xb[far*3+0], y = xb[far*3+1], z = xb[far*3+2];
      float* nz = new_xyz + ((size_t)b * npoint + it) * 3;
      nz[0] = x; nz[1] = y; nz[2] = z;
      cpt[0] = x; cpt[1] = y; cpt[2] = z;
    }
    __syncthreads();
    const float cx = cpt[0], cy = cpt[1], cz = cpt[2];
    float bv = -1.0f; int bi = 0x7fffffff;
    for (int p = 0; p < P; ++p) {
      int i = tid + (p << 8);
      if (i < N) {
        float dx = xb[i*3+0]-cx, dy = xb[i*3+1]-cy, dz = xb[i*3+2]-cz;
        float d = dx*dx + dy*dy + dz*dz;
        float m = mind[p];
        m = d < m ? d : m;
        mind[p] = m;
        if (m > bv) { bv = m; bi = i; }
      }
    }
    s_v[tid] = bv; s_i[tid] = bi;
    __syncthreads();
    for (int o = 128; o > 0; o >>= 1) {
      if (tid < o) {
        float v2 = s_v[tid+o]; int i2 = s_i[tid+o];
        if (v2 > s_v[tid] || (v2 == s_v[tid] && i2 < s_i[tid])) { s_v[tid] = v2; s_i[tid] = i2; }
      }
      __syncthreads();
    }
    far = s_i[0];
    __syncthreads();
  }
}

// ---------------------------------------------------------------------------
// Ball query: thread per centroid, first nsample hits in index order,
// pad with first hit (matches reference top_k(-key) construction).
// ---------------------------------------------------------------------------
__global__ void ball_query_kernel(const float* __restrict__ xyz,
                                  const float* __restrict__ new_xyz,
                                  int N, int np, int ns, float r2,
                                  int* __restrict__ idx)
{
  int t = blockIdx.x * blockDim.x + threadIdx.x;
  if (t >= NB * np) return;
  int b = t / np;
  const float* c = new_xyz + (size_t)t * 3;
  float cx = c[0], cy = c[1], cz = c[2];
  const float* xb = xyz + (size_t)b * N * 3;
  int* o = idx + (size_t)t * ns;
  int cnt = 0;
  for (int i = 0; i < N && cnt < ns; ++i) {
    float dx = xb[i*3+0]-cx, dy = xb[i*3+1]-cy, dz = xb[i*3+2]-cz;
    if (dx*dx + dy*dy + dz*dz < r2) o[cnt++] = i;
  }
  int f = (cnt > 0) ? o[0] : 0;
  for (int k = cnt; k < ns; ++k) o[k] = f;
}

// ---------------------------------------------------------------------------
// Grouping for SA1/SA2: build X0 f16 in [B][np*ns][Kpad] layout.
// ch 0..2 = relative xyz, ch 3..3+Cf-1 = gathered feats (or absolute xyz for
// SA1 where feats == xyz), remainder zero-padded up to Kpad.
// feats layout: [B][N][Cf] f16 (transposed activation layout).
// ---------------------------------------------------------------------------
__global__ void group_kernel(const float* __restrict__ xyz,
                             const float* __restrict__ new_xyz,
                             const int* __restrict__ idx,
                             const half_t* __restrict__ feats,
                             int N, int np, int ns, int Cf, int Kpad,
                             half_t* __restrict__ X)
{
  int t = blockIdx.x * blockDim.x + threadIdx.x;
  int total = NB * np * ns;
  if (t >= total) return;
  int s = t % ns;
  int j = (t / ns) % np;
  int b = t / (ns * np);
  int S = np * ns;
  int i = idx[t];
  half_t* xc = X + ((size_t)b * S + (size_t)j * ns + s) * Kpad;
  const float* c = new_xyz + ((size_t)b * np + j) * 3;
  const float* p = xyz + ((size_t)b * N + i) * 3;
  xc[0] = (half_t)(p[0] - c[0]);
  xc[1] = (half_t)(p[1] - c[1]);
  xc[2] = (half_t)(p[2] - c[2]);
  if (feats) {
    const half_t* f = feats + ((size_t)b * N + i) * Cf;
    for (int cc = 0; cc < Cf; ++cc) xc[3+cc] = f[cc];
  } else {
    xc[3] = (half_t)p[0]; xc[4] = (half_t)p[1]; xc[5] = (half_t)p[2];
  }
  for (int cc = 3 + Cf; cc < Kpad; ++cc) xc[cc] = (half_t)0.f;
}

// ---------------------------------------------------------------------------
// Grouping for SA3 (global): absolute xyz + feats, S points, zero-pad K.
// ---------------------------------------------------------------------------
__global__ void group3_kernel(const float* __restrict__ xyz,
                              const half_t* __restrict__ feats,
                              int S, int Cf, int Kpad,
                              half_t* __restrict__ X)
{
  int t = blockIdx.x * blockDim.x + threadIdx.x;
  if (t >= NB * S) return;
  int s = t % S, b = t / S;
  half_t* xc = X + ((size_t)b * S + s) * Kpad;
  const float* p = xyz + ((size_t)b * S + s) * 3;
  xc[0] = (half_t)p[0]; xc[1] = (half_t)p[1]; xc[2] = (half_t)p[2];
  const half_t* f = feats + ((size_t)b * S + s) * Cf;
  for (int cc = 0; cc < Cf; ++cc) xc[3+cc] = f[cc];
  for (int cc = 3 + Cf; cc < Kpad; ++cc) xc[cc] = (half_t)0.f;
}

// ---------------------------------------------------------------------------
// Max over nsample per group. X: [B][np*ns][C] f16 -> out: [B][np_pad][C] f16
// (np_pad >= np; extra rows zeroed — used to pad SA3's single point to a
// full WMMA N-tile).
// ---------------------------------------------------------------------------
__global__ void group_max_kernel(const half_t* __restrict__ X,
                                 half_t* __restrict__ out,
                                 int C, int np, int ns, int np_pad)
{
  int t = blockIdx.x * blockDim.x + threadIdx.x;
  int total = NB * np_pad * C;
  if (t >= total) return;
  int c = t % C;
  int j = (t / C) % np_pad;
  int b = t / (C * np_pad);
  half_t* o = out + ((size_t)b * np_pad + j) * C + c;
  if (j < np) {
    float m = -3.0e38f;
    const half_t* x = X + ((size_t)b * np * ns + (size_t)j * ns) * C + c;
    for (int s = 0; s < ns; ++s) m = fmaxf(m, (float)x[(size_t)s * C]);
    *o = (half_t)m;
  } else {
    *o = (half_t)0.f;
  }
}

// ---------------------------------------------------------------------------
// Conv layer prep: W -> f16 zero-padded to Kpad, and fused eval-BN params:
// scale[m] = gamma*inv, shift[m] = bias*gamma*inv + beta, inv = 1/sqrt(1+eps)
// ---------------------------------------------------------------------------
__global__ void prep_conv_kernel(const float* __restrict__ W,
                                 const float* __restrict__ bias,
                                 const float* __restrict__ gamma,
                                 const float* __restrict__ beta,
                                 half_t* __restrict__ W16,
                                 float* __restrict__ scale,
                                 float* __restrict__ shift,
                                 int M, int Cin, int Kpad)
{
  int t = blockIdx.x * blockDim.x + threadIdx.x;
  if (t < M * Kpad) {
    int m = t / Kpad, k = t % Kpad;
    W16[t] = (k < Cin) ? (half_t)W[(size_t)m * Cin + k] : (half_t)0.f;
  }
  if (t < M) {
    float inv = rsqrtf(1.0f + 1e-5f);
    float s = gamma[t] * inv;
    scale[t] = s;
    shift[t] = bias[t] * s + beta[t];
  }
}

// ---------------------------------------------------------------------------
// 3-layer 16-wide MLP on betas (leaky ReLU between layers). One block,
// thread (b,o) = tid; x ping-pongs through shared memory. B=16, C=16.
// ---------------------------------------------------------------------------
__global__ __launch_bounds__(256)
void ff3_kernel(const float* __restrict__ xin,
                const float* __restrict__ W0, const float* __restrict__ b0,
                const float* __restrict__ W1, const float* __restrict__ b1,
                const float* __restrict__ W2, const float* __restrict__ b2,
                float* __restrict__ out)
{
  __shared__ float xs[256];
  int tid = threadIdx.x;
  int b = tid >> 4, o = tid & 15;
  xs[tid] = xin[tid];
  __syncthreads();
  float y = b0[o];
  for (int i = 0; i < 16; ++i) y += W0[o*16+i] * xs[b*16+i];
  y = y > 0.f ? y : 0.01f * y;
  __syncthreads(); xs[tid] = y; __syncthreads();
  y = b1[o];
  for (int i = 0; i < 16; ++i) y += W1[o*16+i] * xs[b*16+i];
  y = y > 0.f ? y : 0.01f * y;
  __syncthreads(); xs[tid] = y; __syncthreads();
  y = b2[o];
  for (int i = 0; i < 16; ++i) y += W2[o*16+i] * xs[b*16+i];
  out[tid] = y;
}

// ---------------------------------------------------------------------------
// Hyper weight-gen: Wmat[b][r] = dot(h[b], Wn_W[r]) + Wn_b[r]  (f16 out).
// Streams Wn_W exactly once (the 23.3 TB/s-bound stage); h kept in LDS.
// ---------------------------------------------------------------------------
__global__ __launch_bounds__(256)
void wgen_kernel(const float* __restrict__ WnW, const float* __restrict__ Wnb,
                 const float* __restrict__ h, half_t* __restrict__ Wmat, int R)
{
  __shared__ float hs[256];
  hs[threadIdx.x] = h[threadIdx.x];
  __syncthreads();
  int r = blockIdx.x * blockDim.x + threadIdx.x;
  if (r >= R) return;
  float w[16];
  for (int i = 0; i < 16; ++i) w[i] = WnW[(size_t)r * 16 + i];
  float bb = Wnb[r];
  for (int b = 0; b < NB; ++b) {
    float acc = bb;
    for (int i = 0; i < 16; ++i) acc += hs[b*16+i] * w[i];
    Wmat[(size_t)b * R + r] = (half_t)acc;
  }
}

// Hyper bias: bvec[b][m] = dot(g[b], bn_W[m]) + bn_b[m]
__global__ void bvec_kernel(const float* __restrict__ bnW,
                            const float* __restrict__ bnb,
                            const float* __restrict__ g,
                            float* __restrict__ bvec, int fdim)
{
  int t = blockIdx.x * blockDim.x + threadIdx.x;
  if (t >= NB * fdim) return;
  int m = t % fdim, b = t / fdim;
  float acc = bnb[m];
  for (int i = 0; i < 16; ++i) acc += g[b*16+i] * bnW[(size_t)m * 16 + i];
  bvec[t] = acc;
}

// ---------------------------------------------------------------------------
// Generic WMMA GEMM: Y[b][s][m] = act(scale[m]*(sum_k A[b?][m][k]*X[b][s][k])
//                                      + shift[(b?),m])
// A: f16 [Bopt][M][Kpad] (aBatchStride=0 -> shared weights)
// X: f16 [B][S][Kpad], Y: f16 [B][S][M].
// One wave computes a 32M x 64N tile: 2 A-fragments + 4 B-fragments
// (B reused across both m-tiles) -> 8 v_wmma_f32_16x16x32_f16 per K-step.
// Fragment packing used here (contiguous K, same order for A and B):
//   A 16x32: lane&15 = row M; lanes 0-15 hold K 0..15, lanes 16-31 K 16..31
//   B 32x16: lane&15 = col N; lanes 0-15 hold K 0..15, lanes 16-31 K 16..31
//   C 16x16: lane&15 = col N; vreg r -> row ((lane>=16)?8:0)+r
// ---------------------------------------------------------------------------
__global__ __launch_bounds__(32)
void wmma_gemm_f16(const half_t* __restrict__ A, const half_t* __restrict__ X,
                   half_t* __restrict__ Y,
                   const float* __restrict__ scale, const float* __restrict__ shift,
                   int M, int Kpad, int S,
                   unsigned long long aBatchStride, int shiftBatched, int act)
{
  const int lane = threadIdx.x;
  const int nrow = lane & 15;
  const int hi   = lane >> 4;
  const int s0 = blockIdx.x * 64;
  const int m0 = blockIdx.y * 32;
  const int b  = blockIdx.z;

  const half_t* Ab0 = A + (size_t)b * aBatchStride + (size_t)(m0 + nrow)      * Kpad;
  const half_t* Ab1 = A + (size_t)b * aBatchStride + (size_t)(m0 + 16 + nrow) * Kpad;
  const half_t* Xb = X + (size_t)b * S * Kpad;
  const half_t* x0 = Xb + (size_t)(s0 +  0 + nrow) * Kpad + hi * 16;
  const half_t* x1 = Xb + (size_t)(s0 + 16 + nrow) * Kpad + hi * 16;
  const half_t* x2 = Xb + (size_t)(s0 + 32 + nrow) * Kpad + hi * 16;
  const half_t* x3 = Xb + (size_t)(s0 + 48 + nrow) * Kpad + hi * 16;
  const int kb = hi * 16;

  v8f c00 = {}, c01 = {}, c02 = {}, c03 = {};
  v8f c10 = {}, c11 = {}, c12 = {}, c13 = {};
  for (int k0 = 0; k0 < Kpad; k0 += 32) {
    v8h a0lo = *(const v8h*)(Ab0 + k0 + kb);
    v8h a0hi = *(const v8h*)(Ab0 + k0 + 8 + kb);
    v8h a1lo = *(const v8h*)(Ab1 + k0 + kb);
    v8h a1hi = *(const v8h*)(Ab1 + k0 + 8 + kb);
    v16h a0, a1;
#pragma unroll
    for (int j = 0; j < 8; ++j) {
      a0[j] = a0lo[j]; a0[8+j] = a0hi[j];
      a1[j] = a1lo[j]; a1[8+j] = a1hi[j];
    }
    v16h b0 = *(const v16h*)(x0 + k0);
    v16h b1 = *(const v16h*)(x1 + k0);
    v16h b2 = *(const v16h*)(x2 + k0);
    v16h b3 = *(const v16h*)(x3 + k0);
    c00 = __builtin_amdgcn_wmma_f32_16x16x32_f16(false, a0, false, b0, (short)0, c00, false, false);
    c01 = __builtin_amdgcn_wmma_f32_16x16x32_f16(false, a0, false, b1, (short)0, c01, false, false);
    c02 = __builtin_amdgcn_wmma_f32_16x16x32_f16(false, a0, false, b2, (short)0, c02, false, false);
    c03 = __builtin_amdgcn_wmma_f32_16x16x32_f16(false, a0, false, b3, (short)0, c03, false, false);
    c10 = __builtin_amdgcn_wmma_f32_16x16x32_f16(false, a1, false, b0, (short)0, c10, false, false);
    c11 = __builtin_amdgcn_wmma_f32_16x16x32_f16(false, a1, false, b1, (short)0, c11, false, false);
    c12 = __builtin_amdgcn_wmma_f32_16x16x32_f16(false, a1, false, b2, (short)0, c12, false, false);
    c13 = __builtin_amdgcn_wmma_f32_16x16x32_f16(false, a1, false, b3, (short)0, c13, false, false);
  }

  v8f cc[2][4] = {{c00, c01, c02, c03}, {c10, c11, c12, c13}};
#pragma unroll
  for (int mt = 0; mt < 2; ++mt) {
    const int mrow = m0 + mt * 16 + hi * 8;
    float sc[8], sh[8];
#pragma unroll
    for (int r = 0; r < 8; ++r) {
      int m = mrow + r;
      sc[r] = scale ? scale[m] : 1.0f;
      sh[r] = shift ? (shiftBatched ? shift[(size_t)b * M + m] : shift[m]) : 0.0f;
    }
#pragma unroll
    for (int tt = 0; tt < 4; ++tt) {
      half_t* yc = Y + ((size_t)b * S + s0 + tt*16 + nrow) * M + mrow;
#pragma unroll
      for (int r = 0; r < 8; ++r) {
        float y = cc[mt][tt][r] * sc[r] + sh[r];
        if (act == 1)      y = y > 0.f ? y : 0.f;
        else if (act == 2) y = y > 0.f ? y : 0.01f * y;
        yc[r] = (half_t)y;
      }
    }
  }
}

// Final FC: out[b][o] = sum_f feats[b][0][f] * fcW[o][f] + fcb[o]
// feats: [B][64][1024] f16 (only column 0 valid).
__global__ void final_fc_kernel(const half_t* __restrict__ feats,
                                const float* __restrict__ fcW,
                                const float* __restrict__ fcb,
                                float* __restrict__ out)
{
  int t = blockIdx.x * blockDim.x + threadIdx.x;
  if (t >= NB * 256) return;
  int o = t % 256, b = t / 256;
  const half_t* f = feats + (size_t)b * 64 * 1024;
  const float* w = fcW + (size_t)o * 1024;
  float acc = fcb[o];
  for (int k = 0; k < 1024; ++k) acc += (float)f[k] * w[k];
  out[t] = acc;
}

// ---------------------------------------------------------------------------
// Host orchestration
// Input flattening order (dict order, nested in construction order):
//   0: pointcloud  1: betas
//   2..37:  9 conv layers x (W, b, gamma, beta)
//   38..181: 3 film modules x 3 layers x (hW0,hb0,hW1,hb1,hW2,hb2,
//                                         gW0,gb0,gW1,gb1,gW2,gb2,
//                                         Wn_W, Wn_b, bn_W, bn_b)
//   182: fc_W  183: fc_b
// ---------------------------------------------------------------------------
extern "C" void kernel_launch(void* const* d_in, const int* in_sizes, int n_in,
                              void* d_out, int out_size, void* d_ws, size_t ws_size,
                              hipStream_t stream)
{
  (void)in_sizes; (void)out_size; (void)ws_size;
  if (n_in < 184) return;

  const float* pc    = (const float*)d_in[0];
  const float* betas = (const float*)d_in[1];
  auto SAW = [&](int l, int k) { return (const float*)d_in[2 + 4*l + k]; };
  auto FP  = [&](int m, int li, int k) { return (const float*)d_in[38 + 48*m + 16*li + k]; };
  const float* fcW = (const float*)d_in[182];
  const float* fcb = (const float*)d_in[183];

  char* wsb = (char*)d_ws;
  size_t off = 0;
  auto alloc = [&](size_t bytes) -> void* {
    size_t o = off;
    off += (bytes + 255) & ~(size_t)255;
    return (void*)(wsb + o);
  };
  int*    fidx1 = (int*)   alloc((size_t)NB*512*sizeof(int));
  float*  nx1   = (float*) alloc((size_t)NB*512*3*sizeof(float));
  int*    idx1  = (int*)   alloc((size_t)NB*512*64*sizeof(int));
  int*    fidx2 = (int*)   alloc((size_t)NB*256*sizeof(int));
  float*  nx2   = (float*) alloc((size_t)NB*256*3*sizeof(float));
  int*    idx2  = (int*)   alloc((size_t)NB*256*64*sizeof(int));
  float*  scaleb= (float*) alloc(1024*sizeof(float));
  float*  shiftb= (float*) alloc(1024*sizeof(float));
  half_t* W16   = (half_t*)alloc((size_t)1024*512*sizeof(half_t));
  float*  hbuf  = (float*) alloc(256*sizeof(float));
  float*  gbuf  = (float*) alloc(256*sizeof(float));
  float*  bvecb = (float*) alloc((size_t)NB*1024*sizeof(float));
  half_t* XF1   = (half_t*)alloc((size_t)1048576*sizeof(half_t));  // 2 MiB
  half_t* XF2   = (half_t*)alloc((size_t)1048576*sizeof(half_t));  // 2 MiB
  half_t* XA    = (half_t*)alloc((size_t)67108864*sizeof(half_t)); // 128 MiB
  half_t* XB    = (half_t*)alloc((size_t)41943040*sizeof(half_t)); // 80 MiB
  half_t* Wmat  = XB;  // overlay: hyper W is generated only when XB is dead

  auto conv = [&](int l, const half_t* Xin, half_t* Xout,
                  int M, int Cin, int Kpad, int S) {
    prep_conv_kernel<<<(M*Kpad + 255)/256, 256, 0, stream>>>(
        SAW(l,0), SAW(l,1), SAW(l,2), SAW(l,3), W16, scaleb, shiftb, M, Cin, Kpad);
    wmma_gemm_f16<<<dim3(S/64, M/32, NB), 32, 0, stream>>>(
        W16, Xin, Xout, scaleb, shiftb, M, Kpad, S, 0ULL, 0, /*relu*/1);
  };

  auto hyper = [&](int m, int fdim, int S) {
    half_t* xin = XF1;  // feats start in XF1, end in XF2 after 3 layers
    half_t* xout = XF2;
    int R = fdim * fdim;
    for (int li = 0; li < 3; ++li) {
      ff3_kernel<<<1, 256, 0, stream>>>(betas,
          FP(m,li,0), FP(m,li,1), FP(m,li,2), FP(m,li,3), FP(m,li,4), FP(m,li,5), hbuf);
      ff3_kernel<<<1, 256, 0, stream>>>(betas,
          FP(m,li,6), FP(m,li,7), FP(m,li,8), FP(m,li,9), FP(m,li,10), FP(m,li,11), gbuf);
      wgen_kernel<<<(R + 255)/256, 256, 0, stream>>>(FP(m,li,12), FP(m,li,13), hbuf, Wmat, R);
      bvec_kernel<<<(NB*fdim + 255)/256, 256, 0, stream>>>(FP(m,li,14), FP(m,li,15), gbuf, bvecb, fdim);
      int act = (li < 2) ? 2 : 0;  // leaky between layers, none after last
      wmma_gemm_f16<<<dim3(S/64, fdim/32, NB), 32, 0, stream>>>(
          Wmat, xin, xout, (const float*)nullptr, bvecb, fdim, fdim, S,
          (unsigned long long)R, /*shiftBatched*/1, act);
      half_t* tmp = xin; xin = xout; xout = tmp;
    }
  };

  // ----------------- SA module 1: np=512, r=0.2, ns=64, MLP [6,64,64,128]
  fps_kernel<<<NB, 256, 0, stream>>>(pc, 4096, 512, fidx1, nx1);
  ball_query_kernel<<<(NB*512 + 255)/256, 256, 0, stream>>>(pc, nx1, 4096, 512, 64, 0.04f, idx1);
  group_kernel<<<((size_t)NB*512*64 + 255)/256, 256, 0, stream>>>(
      pc, nx1, idx1, (const half_t*)nullptr, 4096, 512, 64, 3, 32, XB);
  conv(0, XB, XA,  64,  6, 32, 32768);
  conv(1, XA, XB,  64, 64, 64, 32768);
  conv(2, XB, XA, 128, 64, 64, 32768);
  group_max_kernel<<<((size_t)NB*512*128 + 255)/256, 256, 0, stream>>>(XA, XF1, 128, 512, 64, 512);
  hyper(0, 128, 512);   // feats1 -> XF2: [B][512][128]

  // ----------------- SA module 2: np=256, r=0.4, ns=64, MLP [131,128,128,256]
  fps_kernel<<<NB, 256, 0, stream>>>(nx1, 512, 256, fidx2, nx2);
  ball_query_kernel<<<(NB*256 + 255)/256, 256, 0, stream>>>(nx1, nx2, 512, 256, 64, 0.16f, idx2);
  group_kernel<<<((size_t)NB*256*64 + 255)/256, 256, 0, stream>>>(
      nx1, nx2, idx2, XF2, 512, 256, 64, 128, 160, XB);
  conv(3, XB, XA, 128, 131, 160, 16384);
  conv(4, XA, XB, 128, 128, 128, 16384);
  conv(5, XB, XA, 256, 128, 128, 16384);
  group_max_kernel<<<((size_t)NB*256*256 + 255)/256, 256, 0, stream>>>(XA, XF1, 256, 256, 64, 256);
  hyper(1, 256, 256);   // feats2 -> XF2: [B][256][256]

  // ----------------- SA module 3: global group, MLP [259,256,512,1024]
  group3_kernel<<<(NB*256 + 255)/256, 256, 0, stream>>>(nx2, XF2, 256, 256, 288, XB);
  conv(6, XB, XA,  256, 259, 288, 256);
  conv(7, XA, XB,  512, 256, 256, 256);
  conv(8, XB, XA, 1024, 512, 512, 256);
  // max over all 256 points -> one column, padded to a 64-wide N tile
  group_max_kernel<<<((size_t)NB*64*1024 + 255)/256, 256, 0, stream>>>(XA, XF1, 1024, 1, 256, 64);
  hyper(2, 1024, 64);   // feats3 -> XF2: [B][64][1024], column 0 valid

  // ----------------- Final FC 1024 -> 256
  final_fc_kernel<<<(NB*256 + 255)/256, 256, 0, stream>>>(XF2, fcW, fcb, (float*)d_out);
}
